// Decoder_61435212202038
// MI455X (gfx1250) — compile-verified, matrix operand order checked by reading, not run on hardware
//
#include <hip/hip_runtime.h>
#include <hip/hip_bf16.h>
#include <math.h>

// ---------------------------------------------------------------------------
// Tacotron2 decoder forward for MI455X (gfx1250, wave32, WMMA).
//   B=32, T_ENC=256, T_DEC=512, N_MELS=80, ENC_DIM=512, PRENET=256,
//   ARNN=DRNN=1024, ATT_DIM=128, LOC_F=32, LOC_K=31.
// Strategy:
//   * hoist prenet (teacher-forced, recurrence-independent) into 2 big WMMA
//     GEMMs over all 512 steps; hoist keys = memory @ Wm.T.
//   * per step: attLSTM (WMMA, 16 WGs, fused gates via LDS) -> attention
//     (f32 VALU, 1 WG per batch) -> decLSTM (WMMA) -> output projection.
//   * weights pre-converted to f16 (L2-resident across all 512 steps).
// Workspace use: ~61 MB.
// ---------------------------------------------------------------------------

typedef _Float16 h8   __attribute__((ext_vector_type(8)));
typedef _Float16 v16h __attribute__((ext_vector_type(16)));
typedef float    v8f  __attribute__((ext_vector_type(8)));

#define NB    32      // batch
#define TENC  256
#define TDEC  512
#define NMELS 80
#define EDIM  512
#define PRE   256
#define HID   1024
#define ADIM  128

static constexpr size_t GATE_OFF  = (size_t)NB * NMELS * TDEC;          // 1310720
static constexpr size_t ALIGN_OFF = GATE_OFF + (size_t)NB * TDEC;       // 1327104

// ---------------------------------------------------------------- WMMA utils
__device__ __forceinline__ v8f wmma_f16(v16h a, v16h b, v8f c) {
  // v_wmma_f32_16x16x32_f16 : D = A(16x32 f16) * B(32x16 f16) + C(16x16 f32)
  return __builtin_amdgcn_wmma_f32_16x16x32_f16(false, a, false, b, (short)0, c,
                                                false, false);
}

// A-fragment (16x32, f16), lane = row within the 16-row tile.
// Per ISA 7.12.2: lanes 0-15 hold K {k0..k0+7, k0+16..k0+23}; lanes 16-31 hold
// K {k0+8..k0+15, k0+24..k0+31}.  B-fragment uses the dual mapping with
// lane = output column; since weights are stored [N][K] row-major we can use
// the same loader with row0 = column base.
__device__ __forceinline__ v16h load_frag(const _Float16* __restrict__ base,
                                          int row0, int ld, int k0, int lane) {
  const int r  = row0 + (lane & 15);
  const int ko = k0 + ((lane & 16) ? 8 : 0);
  const _Float16* p = base + (size_t)r * ld + ko;
  h8 lo = *(const h8*)(p);
  h8 hi = *(const h8*)(p + 16);
  v16h f;
#pragma unroll
  for (int i = 0; i < 8; ++i) { f[i] = lo[i]; f[i + 8] = hi[i]; }
  return f;
}

__device__ __forceinline__ float sigf(float x) { return 1.0f / (1.0f + __expf(-x)); }

// deterministic dropout mask (reference hardcodes training dropout; JAX RNG is
// not reproducible here, so use a fixed integer hash -> {0, 2} keep-scale 0.5)
__device__ __forceinline__ float dropmask(int layer, int t, int m, int u) {
  uint32_t x = (uint32_t)layer * 0x9E3779B1u ^ (uint32_t)t * 0x85EBCA6Bu ^
               (uint32_t)m * 0xC2B2AE35u ^ (uint32_t)u * 0x27D4EB2Fu;
  x ^= x >> 16; x *= 0x7FEB352Du; x ^= x >> 15; x *= 0x846CA68Bu; x ^= x >> 16;
  return (x & 1u) ? 2.0f : 0.0f;
}

// -------------------------------------------------------------- precompute
__global__ __launch_bounds__(256) void convert_f16_kernel(
    const float* __restrict__ src, _Float16* __restrict__ dst,
    int N, int K, int Kp) {
  int idx = blockIdx.x * 256 + threadIdx.x;
  if (idx >= N * Kp) return;
  int n = idx / Kp, k = idx - n * Kp;
  dst[idx] = (k < K) ? (_Float16)src[(size_t)n * K + k] : (_Float16)0.0f;
}

// dec_inputs[t][m][k] (K padded 80->96), teacher forced: t==0 -> GO frame (0)
__global__ __launch_bounds__(256) void decin_kernel(
    const float* __restrict__ mel_target, _Float16* __restrict__ decin) {
  int idx = blockIdx.x * 256 + threadIdx.x;
  if (idx >= TDEC * NB * 96) return;
  int t = idx / (NB * 96);
  int r = idx - t * NB * 96;
  int m = r / 96, k = r - m * 96;
  float v = 0.0f;
  if (k < NMELS && t > 0)
    v = mel_target[(size_t)m * NMELS * TDEC + (size_t)k * TDEC + (t - 1)];
  decin[idx] = (_Float16)v;
}

// keys[b*TENC+t][ADIM] = memory[b,t,:] @ Wm.T   (one-time, f32 VALU)
__global__ __launch_bounds__(256) void keys_kernel(
    const float* __restrict__ memory, const float* __restrict__ Wm,
    float* __restrict__ keys) {
  int idx = blockIdx.x * 256 + threadIdx.x;   // 8192*128 outputs
  int row = idx >> 7, a = idx & 127;
  const float* mrow = memory + (size_t)row * EDIM;
  const float* wrow = Wm + (size_t)a * EDIM;
  float s = 0.0f;
  for (int k = 0; k < EDIM; ++k) s += mrow[k] * wrow[k];
  keys[(size_t)row * ADIM + a] = s;
}

// prenet layer (all 512 steps batched): Y = f16(relu(X@W.T + b) * drop)
// X:[TDEC*32][Kp] f16, W:[256][Kp] f16, Y:[TDEC*32][256] f16.  1 WG per step.
__global__ __launch_bounds__(256) void prenet_gemm_kernel(
    const _Float16* __restrict__ X, const _Float16* __restrict__ W,
    const float* __restrict__ bias, _Float16* __restrict__ Y,
    int Kp, int layer) {
  const int t = blockIdx.x;
  const int lane = threadIdx.x & 31, w = threadIdx.x >> 5;
  const _Float16* Xr = X + (size_t)t * NB * Kp;
  const int n0 = w * 32;

  v8f acc[2][2];
#pragma unroll
  for (int i = 0; i < 2; ++i)
#pragma unroll
    for (int j = 0; j < 2; ++j) acc[i][j] = (v8f)(0.0f);

  for (int k0 = 0; k0 < Kp; k0 += 32) {
    v16h a0 = load_frag(Xr, 0, Kp, k0, lane);
    v16h a1 = load_frag(Xr, 16, Kp, k0, lane);
    v16h b0 = load_frag(W, n0, Kp, k0, lane);
    v16h b1 = load_frag(W, n0 + 16, Kp, k0, lane);
    acc[0][0] = wmma_f16(a0, b0, acc[0][0]);
    acc[0][1] = wmma_f16(a0, b1, acc[0][1]);
    acc[1][0] = wmma_f16(a1, b0, acc[1][0]);
    acc[1][1] = wmma_f16(a1, b1, acc[1][1]);
  }
#pragma unroll
  for (int i = 0; i < 2; ++i)
#pragma unroll
    for (int j = 0; j < 2; ++j) {
      int n = n0 + j * 16 + (lane & 15);
      int mb = i * 16 + ((lane & 16) ? 8 : 0);
      float bv = bias[n];
#pragma unroll
      for (int r = 0; r < 8; ++r) {
        int m = mb + r;
        float v = fmaxf(acc[i][j][r] + bv, 0.0f) * dropmask(layer, t, m, n);
        Y[((size_t)t * NB + m) * PRE + n] = (_Float16)v;
      }
    }
}

// ------------------------------------------------------------- LSTM (WMMA)
// One step of an LSTM cell, hidden=1024. grid=16 WGs, each WG owns a 64-unit
// hidden slice and computes all 4 gates for it (8 waves, each a 32x32 z-tile).
// Input = concat(xa[32][Ka], xb[32][Kb]) via Wih[4096][Ka+Kb]; hidden hin via
// Whh[4096][Kh].  z tiles exchanged through LDS, gates fused in-kernel.
__global__ __launch_bounds__(256) void lstm_step_kernel(
    const _Float16* __restrict__ xa, int lda, int Ka,
    const _Float16* __restrict__ xb, int ldb, int Kb,
    const _Float16* __restrict__ hin, int Kh,
    const _Float16* __restrict__ Wih, int Kih,
    const _Float16* __restrict__ Whh,
    const float* __restrict__ bih, const float* __restrict__ bhh,
    float* __restrict__ cst, float* __restrict__ hout32,
    _Float16* __restrict__ hout16) {
  __shared__ float zlds[NB * 256];
  const int tid = threadIdx.x, lane = tid & 31, w = tid >> 5;
  const int g = w >> 1;                      // gate 0..3 (i,f,g,o)
  const int hbase = blockIdx.x * 64;         // hidden slice base
  const int nrow0 = g * HID + hbase + (w & 1) * 32;  // weight-row base (32 cols)

  v8f acc[2][2];
#pragma unroll
  for (int i = 0; i < 2; ++i)
#pragma unroll
    for (int j = 0; j < 2; ++j) acc[i][j] = (v8f)(0.0f);

  // segment A of the input
  for (int k0 = 0; k0 < Ka; k0 += 32) {
    v16h a0 = load_frag(xa, 0, lda, k0, lane);
    v16h a1 = load_frag(xa, 16, lda, k0, lane);
    v16h b0 = load_frag(Wih, nrow0, Kih, k0, lane);
    v16h b1 = load_frag(Wih, nrow0 + 16, Kih, k0, lane);
    acc[0][0] = wmma_f16(a0, b0, acc[0][0]);
    acc[0][1] = wmma_f16(a0, b1, acc[0][1]);
    acc[1][0] = wmma_f16(a1, b0, acc[1][0]);
    acc[1][1] = wmma_f16(a1, b1, acc[1][1]);
  }
  // segment B of the input (columns Ka..Ka+Kb of Wih)
  for (int k0 = 0; k0 < Kb; k0 += 32) {
    v16h a0 = load_frag(xb, 0, ldb, k0, lane);
    v16h a1 = load_frag(xb, 16, ldb, k0, lane);
    v16h b0 = load_frag(Wih, nrow0, Kih, Ka + k0, lane);
    v16h b1 = load_frag(Wih, nrow0 + 16, Kih, Ka + k0, lane);
    acc[0][0] = wmma_f16(a0, b0, acc[0][0]);
    acc[0][1] = wmma_f16(a0, b1, acc[0][1]);
    acc[1][0] = wmma_f16(a1, b0, acc[1][0]);
    acc[1][1] = wmma_f16(a1, b1, acc[1][1]);
  }
  // recurrent term
  for (int k0 = 0; k0 < Kh; k0 += 32) {
    v16h a0 = load_frag(hin, 0, Kh, k0, lane);
    v16h a1 = load_frag(hin, 16, Kh, k0, lane);
    v16h b0 = load_frag(Whh, nrow0, Kh, k0, lane);
    v16h b1 = load_frag(Whh, nrow0 + 16, Kh, k0, lane);
    acc[0][0] = wmma_f16(a0, b0, acc[0][0]);
    acc[0][1] = wmma_f16(a0, b1, acc[0][1]);
    acc[1][0] = wmma_f16(a1, b0, acc[1][0]);
    acc[1][1] = wmma_f16(a1, b1, acc[1][1]);
  }

  // z exchange: local column = gate*64 + hl
#pragma unroll
  for (int i = 0; i < 2; ++i)
#pragma unroll
    for (int j = 0; j < 2; ++j) {
      int n = w * 32 + j * 16 + (lane & 15);
      int mb = i * 16 + ((lane & 16) ? 8 : 0);
#pragma unroll
      for (int r = 0; r < 8; ++r) zlds[(mb + r) * 256 + n] = acc[i][j][r];
    }
  __syncthreads();

  // fused gating: 32 batch x 64 hidden = 2048 cells, 8 per thread
#pragma unroll
  for (int ii = 0; ii < 8; ++ii) {
    int f = tid + 256 * ii;
    int m = f >> 6, hl = f & 63;
    int Hg = hbase + hl;
    float zi = zlds[m * 256 + hl]       + bih[Hg]            + bhh[Hg];
    float zf = zlds[m * 256 + 64 + hl]  + bih[HID + Hg]      + bhh[HID + Hg];
    float zg = zlds[m * 256 + 128 + hl] + bih[2 * HID + Hg]  + bhh[2 * HID + Hg];
    float zo = zlds[m * 256 + 192 + hl] + bih[3 * HID + Hg]  + bhh[3 * HID + Hg];
    float co = cst[(size_t)m * HID + Hg];
    float cn = sigf(zf) * co + sigf(zi) * tanhf(zg);
    float hn = sigf(zo) * tanhf(cn);
    cst[(size_t)m * HID + Hg] = cn;
    hout32[(size_t)m * HID + Hg] = hn;
    hout16[(size_t)m * HID + Hg] = (_Float16)hn;
  }
}

// --------------------------------------------------- location attention step
// one WG (256 threads) per batch element; all f32.
__global__ __launch_bounds__(256) void attention_kernel(
    const float* __restrict__ memory, const int* __restrict__ memlen,
    const float* __restrict__ keys, const float* __restrict__ Wq,
    const float* __restrict__ Wv, const float* __restrict__ W_loc,
    const float* __restrict__ b_loc, const float* __restrict__ W_locd,
    const float* __restrict__ ah32, float* __restrict__ aw,
    float* __restrict__ awc, float* __restrict__ ctx32,
    _Float16* __restrict__ ctx16, float* __restrict__ align_out, int step) {
  __shared__ float s_wlocd[ADIM * 32];   // 16 KB
  __shared__ float s_wloc[32 * 62];
  __shared__ float s_wv[ADIM];
  __shared__ float s_bloc[32];
  __shared__ float s_q[ADIM];
  __shared__ float s_awold[TENC];
  __shared__ float s_awcold[TENC];
  __shared__ float s_red[TENC];
  __shared__ float s_awn[TENC];

  const int b = blockIdx.x, tid = threadIdx.x;
  for (int i = tid; i < ADIM * 32; i += 256) s_wlocd[i] = W_locd[i];
  for (int i = tid; i < 32 * 62; i += 256) s_wloc[i] = W_loc[i];
  if (tid < ADIM) s_wv[tid] = Wv[tid];
  if (tid < 32) s_bloc[tid] = b_loc[tid];
  s_awold[tid] = aw[(size_t)b * TENC + tid];
  s_awcold[tid] = awc[(size_t)b * TENC + tid];
  if (tid < ADIM) {                       // query projection q = ah @ Wq.T
    const float* h = ah32 + (size_t)b * HID;
    const float* wr = Wq + (size_t)tid * HID;
    float s = 0.0f;
    for (int k = 0; k < HID; ++k) s += h[k] * wr[k];
    s_q[tid] = s;
  }
  __syncthreads();

  // location conv (pad 15, K=31, 2 channels -> 32 filters) at position t=tid
  float locf[32];
#pragma unroll
  for (int f = 0; f < 32; ++f) locf[f] = s_bloc[f];
  for (int kk = 0; kk < 31; ++kk) {
    int idx = tid + kk - 15;
    if (idx >= 0 && idx < TENC) {
      float a0 = s_awold[idx], a1 = s_awcold[idx];
#pragma unroll
      for (int f = 0; f < 32; ++f)
        locf[f] += s_wloc[f * 62 + kk] * a0 + s_wloc[f * 62 + 31 + kk] * a1;
    }
  }
  // energies: e_t = sum_a tanh(q_a + key_a + locproj_a) * Wv_a
  const float* keyrow = keys + ((size_t)b * TENC + tid) * ADIM;
  float e = 0.0f;
  for (int a = 0; a < ADIM; ++a) {
    float lp = 0.0f;
#pragma unroll
    for (int f = 0; f < 32; ++f) lp += locf[f] * s_wlocd[a * 32 + f];
    e += tanhf(s_q[a] + keyrow[a] + lp) * s_wv[a];
  }
  if (tid >= memlen[b]) e = -1e30f;       // mask padded positions

  // softmax over T_ENC (== blockDim)
  s_red[tid] = e;
  __syncthreads();
  for (int s = 128; s > 0; s >>= 1) {
    if (tid < s) s_red[tid] = fmaxf(s_red[tid], s_red[tid + s]);
    __syncthreads();
  }
  float mx = s_red[0];
  __syncthreads();
  float p = __expf(e - mx);
  s_red[tid] = p;
  __syncthreads();
  for (int s = 128; s > 0; s >>= 1) {
    if (tid < s) s_red[tid] += s_red[tid + s];
    __syncthreads();
  }
  float a_t = p / s_red[0];
  s_awn[tid] = a_t;
  aw[(size_t)b * TENC + tid] = a_t;
  awc[(size_t)b * TENC + tid] = s_awcold[tid] + a_t;
  align_out[((size_t)b * TDEC + step) * TENC + tid] = a_t;
  __syncthreads();

  // context: ctx_d = sum_t aw_t * memory[b,t,d]   (512 dims, 2 per thread)
  for (int d = tid; d < EDIM; d += 256) {
    const float* mp = memory + ((size_t)b * TENC) * EDIM + d;
    float s = 0.0f;
    for (int t = 0; t < TENC; ++t) s += s_awn[t] * mp[(size_t)t * EDIM];
    ctx32[(size_t)b * EDIM + d] = s;
    ctx16[(size_t)b * EDIM + d] = (_Float16)s;
  }
}

// ------------------------------------------------------- output projection
__global__ __launch_bounds__(96) void output_kernel(
    const float* __restrict__ dh, const float* __restrict__ ctx,
    const float* __restrict__ W_lin, const float* __restrict__ b_lin,
    const float* __restrict__ W_gate, const float* __restrict__ b_gate,
    float* __restrict__ out, int step) {
  const int b = blockIdx.x, n = threadIdx.x;
  if (n > NMELS) return;                  // 80 mel rows + 1 gate row
  const float* wr = (n < NMELS) ? (W_lin + (size_t)n * 1536) : W_gate;
  float s = (n < NMELS) ? b_lin[n] : b_gate[0];
  const float* hv = dh + (size_t)b * HID;
  for (int k = 0; k < HID; ++k) s += hv[k] * wr[k];
  const float* cv = ctx + (size_t)b * EDIM;
  for (int k = 0; k < EDIM; ++k) s += cv[k] * wr[HID + k];
  if (n < NMELS) out[((size_t)b * NMELS + n) * TDEC + step] = s;
  else out[GATE_OFF + (size_t)b * TDEC + step] = s;
}

// --------------------------------------------------------------------- host
extern "C" void kernel_launch(void* const* d_in, const int* in_sizes, int n_in,
                              void* d_out, int out_size, void* d_ws,
                              size_t ws_size, hipStream_t stream) {
  const float* memory     = (const float*)d_in[0];
  const float* mel_target = (const float*)d_in[1];
  const int*   memlen     = (const int*)d_in[2];
  const float* W_p1  = (const float*)d_in[3];
  const float* b_p1  = (const float*)d_in[4];
  const float* W_p2  = (const float*)d_in[5];
  const float* b_p2  = (const float*)d_in[6];
  const float* Wih_a = (const float*)d_in[7];
  const float* Whh_a = (const float*)d_in[8];
  const float* bih_a = (const float*)d_in[9];
  const float* bhh_a = (const float*)d_in[10];
  const float* Wq    = (const float*)d_in[11];
  const float* Wm    = (const float*)d_in[12];
  const float* Wv    = (const float*)d_in[13];
  const float* W_loc = (const float*)d_in[14];
  const float* b_loc = (const float*)d_in[15];
  const float* W_locd= (const float*)d_in[16];
  const float* Wih_d = (const float*)d_in[17];
  const float* Whh_d = (const float*)d_in[18];
  const float* bih_d = (const float*)d_in[19];
  const float* bhh_d = (const float*)d_in[20];
  const float* W_lin = (const float*)d_in[21];
  const float* b_lin = (const float*)d_in[22];
  const float* W_gate= (const float*)d_in[23];
  const float* b_gate= (const float*)d_in[24];
  float* out = (float*)d_out;
  char* ws = (char*)d_ws;

  // workspace layout (bytes) ------------------------------------------------
  constexpr size_t SZ_WIH_A16 = 4096ull * 768 * 2;
  constexpr size_t SZ_WHH_A16 = 4096ull * 1024 * 2;
  constexpr size_t SZ_WIH_D16 = 4096ull * 1536 * 2;
  constexpr size_t SZ_WHH_D16 = 4096ull * 1024 * 2;
  constexpr size_t SZ_WP1     = 256ull * 96 * 2;
  constexpr size_t SZ_WP2     = 256ull * 256 * 2;
  constexpr size_t SZ_DECIN   = (size_t)TDEC * NB * 96 * 2;
  constexpr size_t SZ_X       = (size_t)TDEC * NB * PRE * 2;
  constexpr size_t SZ_KEYS    = (size_t)NB * TENC * ADIM * 4;

  size_t o = 0;
  _Float16* wih_a16 = (_Float16*)(ws + o); o += SZ_WIH_A16;
  _Float16* whh_a16 = (_Float16*)(ws + o); o += SZ_WHH_A16;
  _Float16* wih_d16 = (_Float16*)(ws + o); o += SZ_WIH_D16;
  _Float16* whh_d16 = (_Float16*)(ws + o); o += SZ_WHH_D16;
  _Float16* wp1_16  = (_Float16*)(ws + o); o += SZ_WP1;
  _Float16* wp2_16  = (_Float16*)(ws + o); o += SZ_WP2;
  _Float16* decin16 = (_Float16*)(ws + o); o += SZ_DECIN;
  _Float16* x1_16   = (_Float16*)(ws + o); o += SZ_X;
  _Float16* x2_16   = (_Float16*)(ws + o); o += SZ_X;
  float*    keys    = (float*)(ws + o);    o += SZ_KEYS;
  // recurrent state block (zeroed every call, since graph replays reuse ws)
  char* state = ws + o;
  float* ah32  = (float*)(state + 0);
  float* ac    = (float*)(state + 131072);
  float* dh32  = (float*)(state + 262144);
  float* dc    = (float*)(state + 393216);
  float* ctx32 = (float*)(state + 524288);
  float* awb   = (float*)(state + 589824);
  float* awcb  = (float*)(state + 622592);
  _Float16* ah16a = (_Float16*)(state + 655360);
  _Float16* ah16b = (_Float16*)(state + 720896);
  _Float16* dh16a = (_Float16*)(state + 786432);
  _Float16* dh16b = (_Float16*)(state + 851968);
  _Float16* ctx16 = (_Float16*)(state + 917504);
  constexpr size_t STATE_BYTES = 917504 + 32768;   // + ctx16 (32 KB)

  // ---- one-time precompute (re-run every call: deterministic, capturable)
  hipMemsetAsync(state, 0, STATE_BYTES, stream);
  convert_f16_kernel<<<(4096 * 768 + 255) / 256, 256, 0, stream>>>(Wih_a, wih_a16, 4096, 768, 768);
  convert_f16_kernel<<<(4096 * 1024 + 255) / 256, 256, 0, stream>>>(Whh_a, whh_a16, 4096, 1024, 1024);
  convert_f16_kernel<<<(4096 * 1536 + 255) / 256, 256, 0, stream>>>(Wih_d, wih_d16, 4096, 1536, 1536);
  convert_f16_kernel<<<(4096 * 1024 + 255) / 256, 256, 0, stream>>>(Whh_d, whh_d16, 4096, 1024, 1024);
  convert_f16_kernel<<<(256 * 96 + 255) / 256, 256, 0, stream>>>(W_p1, wp1_16, 256, 80, 96);
  convert_f16_kernel<<<(256 * 256 + 255) / 256, 256, 0, stream>>>(W_p2, wp2_16, 256, 256, 256);
  decin_kernel<<<(TDEC * NB * 96 + 255) / 256, 256, 0, stream>>>(mel_target, decin16);
  keys_kernel<<<(NB * TENC * ADIM) / 256, 256, 0, stream>>>(memory, Wm, keys);
  // prenet hoisted over all 512 steps (teacher forcing => no recurrence dep)
  prenet_gemm_kernel<<<TDEC, 256, 0, stream>>>(decin16, wp1_16, b_p1, x1_16, 96, 0);
  prenet_gemm_kernel<<<TDEC, 256, 0, stream>>>(x1_16, wp2_16, b_p2, x2_16, 256, 1);

  // ---- sequential decode loop (stream order provides the per-step sync)
  for (int t = 0; t < TDEC; ++t) {
    const _Float16* ah_in  = (t & 1) ? ah16b : ah16a;
    _Float16*       ah_out = (t & 1) ? ah16a : ah16b;
    const _Float16* dh_in  = (t & 1) ? dh16b : dh16a;
    _Float16*       dh_out = (t & 1) ? dh16a : dh16b;

    // attention LSTM: x = [prenet(t) | ctx], h = ah
    lstm_step_kernel<<<16, 256, 0, stream>>>(
        x2_16 + (size_t)t * NB * PRE, PRE, PRE, ctx16, EDIM, EDIM,
        ah_in, HID, wih_a16, PRE + EDIM, whh_a16, bih_a, bhh_a,
        ac, ah32, ah_out);
    // location-sensitive attention + context
    attention_kernel<<<NB, 256, 0, stream>>>(
        memory, memlen, keys, Wq, Wv, W_loc, b_loc, W_locd,
        ah32, awb, awcb, ctx32, ctx16, out + ALIGN_OFF, t);
    // decoder LSTM: x = [ah | ctx], h = dh
    lstm_step_kernel<<<16, 256, 0, stream>>>(
        ah_out, HID, HID, ctx16, EDIM, EDIM,
        dh_in, HID, wih_d16, HID + EDIM, whh_d16, bih_d, bhh_d,
        dc, dh32, dh_out);
    // mel + gate projection
    output_kernel<<<NB, 96, 0, stream>>>(dh32, ctx32, W_lin, b_lin,
                                         W_gate, b_gate, out, t);
  }
}
